// DoubleConvolutionalEmbedding_44538810860311
// MI455X (gfx1250) — matrix operand order repaired and checked by reading, not compiled, not run on hardware
//
#include <hip/hip_runtime.h>
#include <hip/hip_bf16.h>

typedef __attribute__((ext_vector_type(16))) _Float16 v16h;
typedef __attribute__((ext_vector_type(8)))  _Float16 v8h;
typedef __attribute__((ext_vector_type(8)))  float    v8f;

#define B_   16
#define L1_  65536
#define L2_  131072
#define S_   (L1_ + L2_)
#define C_   16
#define KS_  8
#define E_   256
#define T2_  (L2_ / KS_)        // 16384 last-layer embeddings
#define M_   (B_ * L1_ / KS_)   // 131072 GEMM rows
#define K_   128                // C_*KS_

// ---------------------------------------------------------------------------
// Stage Y: kernel-8/stride-8 conv on last layer -> f16 [B, T2, C]
// ---------------------------------------------------------------------------
__global__ void conv_last_kernel(const int* __restrict__ value,
                                 const int* __restrict__ depth,
                                 const int* __restrict__ pos,
                                 const float* __restrict__ w_v2,
                                 const float* __restrict__ b_v2,
                                 const float* __restrict__ w_d2,
                                 const float* __restrict__ b_d2,
                                 const float* __restrict__ w_p2,
                                 const float* __restrict__ b_p2,
                                 _Float16* __restrict__ Yws) {
  int idx = blockIdx.x * 256 + threadIdx.x;        // B*T2*C = 2^22
  int c  = idx & 15;
  int t2 = (idx >> 4) & (T2_ - 1);
  int b  = idx >> 18;
  float acc = b_v2[c] + b_d2[c] + b_p2[c] + b_p2[16 + c] + b_p2[32 + c];
  size_t base = (size_t)b * S_ + L1_ + (size_t)t2 * KS_;
#pragma unroll
  for (int s = 0; s < KS_; ++s) {
    float vf = (float)value[base + s];
    float df = (float)depth[base + s];
    size_t pb = (base + s) * 3;
    acc += vf * w_v2[c * 8 + s] + df * w_d2[c * 8 + s]
         + (float)pos[pb + 0] * w_p2[c * 8 + s]
         + (float)pos[pb + 1] * w_p2[128 + c * 8 + s]
         + (float)pos[pb + 2] * w_p2[256 + c * 8 + s];
  }
  Yws[idx] = (_Float16)acc;
}

// ---------------------------------------------------------------------------
// Mask prefix-sum machinery (generic rank computation, torch-style ordering)
// ---------------------------------------------------------------------------
__global__ void mask_block_sums_kernel(const int* __restrict__ value,
                                       int* __restrict__ bSums) {
  int bid = blockIdx.x;                 // 0..4095  (b = bid>>8, blk = bid&255)
  int b = bid >> 8, blk = bid & 255;
  int t = blk * 256 + threadIdx.x;
  int m = (value[(size_t)b * S_ + t] == 2) ? 1 : 0;
  __shared__ int sc[256];
  sc[threadIdx.x] = m;
  __syncthreads();
  for (int off = 128; off > 0; off >>= 1) {
    if (threadIdx.x < off) sc[threadIdx.x] += sc[threadIdx.x + off];
    __syncthreads();
  }
  if (threadIdx.x == 0) bSums[bid] = sc[0];
}

__global__ void scan_block_sums_kernel(const int* __restrict__ bSums,
                                       int* __restrict__ bOffs) {
  int b = blockIdx.x;                   // one block per batch
  int tid = threadIdx.x;                // 256 block-sums per batch
  __shared__ int sc[256];
  int v = bSums[b * 256 + tid];
  sc[tid] = v;
  __syncthreads();
  for (int off = 1; off < 256; off <<= 1) {
    int x = (tid >= off) ? sc[tid - off] : 0;
    __syncthreads();
    sc[tid] += x;
    __syncthreads();
  }
  bOffs[b * 256 + tid] = sc[tid] - v;   // exclusive prefix
}

// ---------------------------------------------------------------------------
// Stage X: per-token affine embed, substitute mixed tokens with y[rank] -> f16
// ---------------------------------------------------------------------------
__global__ void build_x_kernel(const int* __restrict__ value,
                               const int* __restrict__ depth,
                               const int* __restrict__ pos,
                               const float* __restrict__ w_v1,
                               const float* __restrict__ b_v1,
                               const float* __restrict__ w_d1,
                               const float* __restrict__ b_d1,
                               const float* __restrict__ w_p1,
                               const float* __restrict__ b_p1,
                               const _Float16* __restrict__ Yws,
                               const int* __restrict__ bOffs,
                               _Float16* __restrict__ Xws) {
  int bid = blockIdx.x;                 // 0..4095
  int b = bid >> 8, blk = bid & 255;
  int tid = threadIdx.x;
  int t = blk * 256 + tid;
  size_t gtok = (size_t)b * S_ + t;
  int v = value[gtok];
  int mixed = (v == 2) ? 1 : 0;
  __shared__ int sc[256];
  sc[tid] = mixed;
  __syncthreads();
  for (int off = 1; off < 256; off <<= 1) {   // inclusive scan
    int x = (tid >= off) ? sc[tid - off] : 0;
    __syncthreads();
    sc[tid] += x;
    __syncthreads();
  }
  __align__(16) _Float16 xr[16];
  if (mixed) {
    int rank = bOffs[bid] + sc[tid] - 1;
    const _Float16* ysrc = Yws + ((size_t)b * T2_ + rank) * C_;
#pragma unroll
    for (int c = 0; c < C_; ++c) xr[c] = ysrc[c];
  } else {
    float vf = (float)v;
    float df = (float)depth[gtok];
    float p0 = (float)pos[gtok * 3 + 0];
    float p1 = (float)pos[gtok * 3 + 1];
    float p2 = (float)pos[gtok * 3 + 2];
#pragma unroll
    for (int c = 0; c < C_; ++c) {
      float xc = vf * w_v1[c] + b_v1[c] + df * w_d1[c] + b_d1[c]
               + p0 * w_p1[c] + p1 * w_p1[16 + c] + p2 * w_p1[32 + c]
               + b_p1[c] + b_p1[16 + c] + b_p1[32 + c];
      xr[c] = (_Float16)xc;
    }
  }
  uint4* dst = (uint4*)(Xws + ((size_t)b * L1_ + t) * C_);
  const uint4* src = (const uint4*)xr;
  dst[0] = src[0];
  dst[1] = src[1];
}

// ---------------------------------------------------------------------------
// Latent GEMM: out[m, e] = sum_k X[m,k]*W[e,k] + b_lat[e]
// M=131072, N=256, K=128 via v_wmma_f32_16x16x32_f16, W staged in LDS
// ---------------------------------------------------------------------------
__global__ void latent_gemm_kernel(const _Float16* __restrict__ X,
                                   const float* __restrict__ w_lat,
                                   const float* __restrict__ b_lat,
                                   float* __restrict__ out) {
  extern __shared__ char smem[];
  _Float16* sW = (_Float16*)smem;                       // [E_][K_] n-major, k-contig
  float* sBias = (float*)(smem + (size_t)E_ * K_ * sizeof(_Float16));
  const int tid = threadIdx.x;

  // Repack w_lat [E,C,KS] (idx e*128 + c*8 + s) -> sW[e*128 + (s*16+c)], f16.
  for (int i = tid; i < E_ * K_; i += 256) {
    int e = i >> 7, r = i & 127;
    int c = r >> 3, s = r & 7;
    sW[(e << 7) + (s << 4) + c] = (_Float16)w_lat[i];
  }
  if (tid < E_) sBias[tid] = b_lat[tid];
  __syncthreads();

  const int lane  = tid & 31;
  const int wave  = tid >> 5;
  const int m0    = blockIdx.x * 128 + wave * 16;       // 8 waves x 16-row M tiles
  const int mrow  = lane & 15;
  const int khalf = lane >> 4;

  union V16 { v16h v; v8h h[2]; };

  // A fragments for all 4 k-steps (16x32 f16 each), §7.12.2 layout:
  //   lanes 0-15: row m, K {k4*32+0..7, k4*32+16..23}; lanes 16-31: +8 offset.
  V16 a[4];
  const _Float16* xrow = X + ((size_t)(m0 + mrow)) * K_ + khalf * 8;
#pragma unroll
  for (int k4 = 0; k4 < 4; ++k4) {
    const v8h* ap = (const v8h*)(xrow + k4 * 32);
    a[k4].h[0] = ap[0];       // K = k4*32 + khalf*8 + {0..7}
    a[k4].h[1] = ap[2];       // K = k4*32 + 16 + khalf*8 + {0..7}
  }

  const int ncol  = lane & 15;
  const int rbase = m0 + 8 * khalf;
#pragma unroll 1
  for (int nt = 0; nt < 16; ++nt) {
    int n = nt * 16 + ncol;
    float bias = sBias[n];
    v8f acc;
#pragma unroll
    for (int j = 0; j < 8; ++j) acc[j] = bias;
#pragma unroll
    for (int k4 = 0; k4 < 4; ++k4) {
      // B fragment (32x16 f16): lane -> col n, K {khalf*8+0..7, 16+khalf*8+0..7}.
      V16 bb;
      const v8h* bp = (const v8h*)(sW + (n << 7) + k4 * 32 + khalf * 8);
      bb.h[0] = bp[0];
      bb.h[1] = bp[2];
      acc = __builtin_amdgcn_wmma_f32_16x16x32_f16(
          /*neg_a=*/false, a[k4].v, /*neg_b=*/false, bb.v,
          /*c_mod=*/(short)0, acc, /*reuse_a=*/false, /*reuse_b=*/false);
    }
    // C/D layout: VGPR j -> row j + 8*(lane>>4), col lane&15.
    float* orow = out + (size_t)rbase * E_ + n;
#pragma unroll
    for (int j = 0; j < 8; ++j) orow[(size_t)j * E_] = acc[j];
  }
}

// ---------------------------------------------------------------------------
extern "C" void kernel_launch(void* const* d_in, const int* in_sizes, int n_in,
                              void* d_out, int out_size, void* d_ws, size_t ws_size,
                              hipStream_t stream) {
  const int*   value = (const int*)d_in[0];
  const int*   depth = (const int*)d_in[1];
  const int*   pos   = (const int*)d_in[2];
  // d_in[3] = len1 (static 65536, baked into constants)
  const float* w_v1  = (const float*)d_in[4];
  const float* b_v1  = (const float*)d_in[5];
  const float* w_d1  = (const float*)d_in[6];
  const float* b_d1  = (const float*)d_in[7];
  const float* w_p1  = (const float*)d_in[8];
  const float* b_p1  = (const float*)d_in[9];
  const float* w_v2  = (const float*)d_in[10];
  const float* b_v2  = (const float*)d_in[11];
  const float* w_d2  = (const float*)d_in[12];
  const float* b_d2  = (const float*)d_in[13];
  const float* w_p2  = (const float*)d_in[14];
  const float* b_p2  = (const float*)d_in[15];
  const float* w_lat = (const float*)d_in[16];
  const float* b_lat = (const float*)d_in[17];
  float* out = (float*)d_out;

  // Workspace layout
  char* ws = (char*)d_ws;
  _Float16* Xws  = (_Float16*)ws;                         // 33,554,432 B
  _Float16* Yws  = (_Float16*)(ws + 33554432);            //  8,388,608 B
  int*      bSum = (int*)(ws + 41943040);                 //     16,384 B
  int*      bOff = (int*)(ws + 41959424);                 //     16,384 B

  // 1) last-layer conv -> Yws (f16)
  conv_last_kernel<<<(B_ * T2_ * C_) / 256, 256, 0, stream>>>(
      value, depth, pos, w_v2, b_v2, w_d2, b_d2, w_p2, b_p2, Yws);

  // 2) per-256-token mask sums, 3) per-batch exclusive scan
  mask_block_sums_kernel<<<B_ * (L1_ / 256), 256, 0, stream>>>(value, bSum);
  scan_block_sums_kernel<<<B_, 256, 0, stream>>>(bSum, bOff);

  // 4) penultimate embed + mixed-token substitution -> Xws (f16)
  build_x_kernel<<<B_ * (L1_ / 256), 256, 0, stream>>>(
      value, depth, pos, w_v1, b_v1, w_d1, b_d1, w_p1, b_p1, Yws, bOff, Xws);

  // 5) latent GEMM with WMMA (128 rows per 256-thread block)
  size_t smem_bytes = (size_t)E_ * K_ * sizeof(_Float16) + E_ * sizeof(float);
  latent_gemm_kernel<<<M_ / 128, 256, smem_bytes, stream>>>(Xws, w_lat, b_lat, out);
}
